// MoeLayers_29755533427274
// MI455X (gfx1250) — matrix operand order repaired and checked by reading, not compiled, not run on hardware
//
#include <hip/hip_runtime.h>
#include <hip/hip_bf16.h>
#include <math.h>

// ---------------------------------------------------------------------------
// MoE layer for MI455X (gfx1250, wave32, WMMA + async global->LDS).
//   D=1024 in, H=4096 hidden, E=8 experts, top-2, T=8192 tokens, fp32 I/O.
// Routed (top-2) compute = 275 GFLOP; weights converted to bf16 once per call
// (134MB -> resident in 192MB L2).  Fused GEMM1 -> GELU -> GEMM2 per
// (expert, 32-token tile) with v_wmma_f32_16x16x32_bf16; B tiles are
// double-buffered and streamed with GLOBAL_LOAD_ASYNC_TO_LDS_B128 so the
// matrix pipes overlap with weight streaming (1 block/WGP -> must self-hide
// latency).
// ---------------------------------------------------------------------------

#define Dc 1024
#define Hc 4096
#define Ec 8
#define Tc 8192
#define BM 32            // token tile
#define BH 256           // H chunk
#define BK1 64           // GEMM1 staged K per pipeline step (2 WMMA K-steps)
#define BK2 32           // GEMM2 staged K per pipeline step
#define HS  (Dc + 8)     // LDS stride of h tile (bf16 elems), 16B-multiple
#define B1S (BK1 + 8)    // 72 elems = 144B rows
#define H1S (BH + 8)     // 264
#define B2S (BK2 + 8)    // 40 elems = 80B rows

typedef __attribute__((ext_vector_type(16))) __bf16 v16bf;
typedef __attribute__((ext_vector_type(8)))  __bf16 v8bf;
typedef __attribute__((ext_vector_type(8)))  float  v8f;

static constexpr size_t B1_TILE = (size_t)BH * B1S;   // elems per B1 buffer
static constexpr size_t B2_TILE = (size_t)Dc * B2S;   // elems per B2 buffer
static constexpr size_t SMEM_BYTES =
    ((size_t)BM * HS + 2 * B1_TILE + (size_t)BM * H1S + 2 * B2_TILE) *
    sizeof(__bf16);                                    // 320,512 B
static_assert(SMEM_BYTES <= 320u * 1024u, "exceeds WGP LDS");

__device__ __forceinline__ v8f wmma_bf16(v16bf a, v16bf b, v8f c) {
  return __builtin_amdgcn_wmma_f32_16x16x32_bf16(
      /*neg_a=*/false, a, /*neg_b=*/false, b,
      /*c_mod=*/(short)0, c, /*reuse_a=*/false, /*reuse_b=*/false);
}

__device__ __forceinline__ v16bf pack16(v8bf lo, v8bf hi) {
  v16bf r;
#pragma unroll
  for (int i = 0; i < 8; ++i) { r[i] = lo[i]; r[8 + i] = hi[i]; }
  return r;
}

// A fragment (16x32): lane m=lane&15 is row M; lane half selects
// K in {0..7,16..23} (lanes 0-15) or {8..15,24..31} (lanes 16-31).
__device__ __forceinline__ v16bf frag_a(const __bf16* row_k0, int lhalf) {
  v8bf lo = *(const v8bf*)(row_k0 + lhalf * 8);
  v8bf hi = *(const v8bf*)(row_k0 + 16 + lhalf * 8);
  return pack16(lo, hi);
}

// B fragment (32x16) from an N-major LDS tile: lane n=lane&15 is column N;
// lanes 0-15 hold K {0..7,16..23}, lanes 16-31 hold K {8..15,24..31}.
__device__ __forceinline__ v16bf frag_b(const __bf16* rowptr, int lhalf) {
  v8bf lo = *(const v8bf*)(rowptr + lhalf * 8);
  v8bf hi = *(const v8bf*)(rowptr + 16 + lhalf * 8);
  return pack16(lo, hi);
}

// ---- CDNA5 async global->LDS (tracked by ASYNCcnt), GVS addressing --------
__device__ __forceinline__ void async_copy_b128(__bf16* lds_dst,
                                                unsigned long long gbase,
                                                unsigned byte_off) {
  unsigned lds_off = (unsigned)(uintptr_t)lds_dst;   // low 32b = LDS offset
  asm volatile("global_load_async_to_lds_b128 %0, %1, %2"
               :
               : "v"(lds_off), "v"(byte_off), "s"(gbase)
               : "memory");
}
__device__ __forceinline__ void wait_async0() {
  asm volatile("s_wait_asynccnt 0" ::: "memory");
}

// Stage 256 rows x 64 K-elems (128B/row) of w1 (N-major) into buf.
__device__ __forceinline__ void issue_b1_tile(unsigned long long gbase,
                                              __bf16* buf, int tid, int k0) {
#pragma unroll
  for (int p = 0; p < 8; ++p) {             // 2048 x 16B segments
    int seg = p * 256 + tid;
    int n = seg >> 3;
    int c = (seg & 7) * 8;
    async_copy_b128(buf + n * B1S + c,
                    gbase, (unsigned)(((n * Dc) + k0 + c) * 2));
  }
}

// Stage 1024 rows x 32 K-elems (64B/row) of w2 (N-major) into buf.
__device__ __forceinline__ void issue_b2_tile(unsigned long long gbase,
                                              __bf16* buf, int tid, int k0abs) {
#pragma unroll
  for (int p = 0; p < 16; ++p) {            // 4096 x 16B segments
    int seg = p * 256 + tid;
    int n = seg >> 2;
    int c = (seg & 3) * 8;
    async_copy_b128(buf + n * B2S + c,
                    gbase, (unsigned)(((n * Hc) + k0abs + c) * 2));
  }
}

// ---------------------------------------------------------------------------
// Kernel 0: zero the (atomically accumulated) output and routing counters.
// ---------------------------------------------------------------------------
__global__ void init_kernel(float* __restrict__ out, int* __restrict__ counts) {
  size_t i = ((size_t)blockIdx.x * blockDim.x + threadIdx.x) * 4;
  float4 z = {0.f, 0.f, 0.f, 0.f};
  *(float4*)(out + i) = z;
  if (blockIdx.x == 0 && threadIdx.x < Ec) counts[threadIdx.x] = 0;
}

// ---------------------------------------------------------------------------
// Kernel 1: per-expert transpose + fp32->bf16: src[e][R][C] -> dst[e][C][R].
// ---------------------------------------------------------------------------
__global__ void transpose_conv_kernel(const float* __restrict__ src,
                                      __bf16* __restrict__ dst, int R, int C) {
  __shared__ float tile[32][33];
  int tilesR = R >> 5, tilesC = C >> 5;
  int b  = blockIdx.x;
  int e  = b / (tilesR * tilesC);
  int rm = b % (tilesR * tilesC);
  int rt = (rm / tilesC) << 5;
  int ct = (rm % tilesC) << 5;
  const float* s = src + (size_t)e * R * C;
  __bf16*      d = dst + (size_t)e * R * C;
  int tx = threadIdx.x & 31, ty = threadIdx.x >> 5;  // 32x8
#pragma unroll
  for (int i = 0; i < 32; i += 8)
    tile[ty + i][tx] = s[(size_t)(rt + ty + i) * C + ct + tx];
  __syncthreads();
#pragma unroll
  for (int i = 0; i < 32; i += 8)
    d[(size_t)(ct + ty + i) * R + rt + tx] = (__bf16)tile[tx][ty + i];
}

// ---------------------------------------------------------------------------
// Kernel 2: gating + LayerNorm + routing.  One wave32 per token.
// ---------------------------------------------------------------------------
__global__ void gate_ln_kernel(const float* __restrict__ x,
                               const float* __restrict__ gate_w,
                               __bf16* __restrict__ xnb,
                               float* __restrict__ comb,
                               int* __restrict__ counts,
                               int* __restrict__ lists) {
  int lane = threadIdx.x & 31;
  int t = blockIdx.x * 8 + (threadIdx.x >> 5);
  const float* xr = x + (size_t)t * Dc;

  float xv[32];
  float sum = 0.f, ssq = 0.f;
  float lg[Ec];
#pragma unroll
  for (int e = 0; e < Ec; ++e) lg[e] = 0.f;

#pragma unroll 4
  for (int i = 0; i < 32; ++i) {
    int d = lane + (i << 5);
    float v = xr[d];
    xv[i] = v;
    sum += v;
    ssq += v * v;
    const float* gw = gate_w + (size_t)d * Ec;
#pragma unroll
    for (int e = 0; e < Ec; ++e) lg[e] += v * gw[e];
  }
#pragma unroll
  for (int off = 16; off; off >>= 1) {
    sum += __shfl_xor(sum, off, 32);
    ssq += __shfl_xor(ssq, off, 32);
#pragma unroll
    for (int e = 0; e < Ec; ++e) lg[e] += __shfl_xor(lg[e], off, 32);
  }
  float mu   = sum * (1.f / Dc);
  float var  = ssq * (1.f / Dc) - mu * mu;
  float rstd = rsqrtf(var + 1e-5f);

  float mx = lg[0];
#pragma unroll
  for (int e = 1; e < Ec; ++e) mx = fmaxf(mx, lg[e]);
  float p[Ec], den = 0.f;
#pragma unroll
  for (int e = 0; e < Ec; ++e) { p[e] = __expf(lg[e] - mx); den += p[e]; }
  float iden = 1.f / den;
#pragma unroll
  for (int e = 0; e < Ec; ++e) p[e] *= iden;

  int s0 = 0;
#pragma unroll
  for (int e = 1; e < Ec; ++e) if (p[e] > p[s0]) s0 = e;
  int s1 = (s0 == 0) ? 1 : 0;
#pragma unroll
  for (int e = 0; e < Ec; ++e) if (e != s0 && p[e] > p[s1]) s1 = e;

  float m2 = fmaxf(p[s0], p[s1]);
  float e0 = __expf(p[s0] - m2), e1 = __expf(p[s1] - m2);
  float inv = 1.f / (e0 + e1);
  float w0 = e0 * inv, w1 = e1 * inv;

  if (lane < Ec)
    comb[(size_t)t * Ec + lane] = (lane == s0) ? w0 : ((lane == s1) ? w1 : 0.f);

#pragma unroll 4
  for (int i = 0; i < 32; ++i) {
    int d = lane + (i << 5);
    xnb[(size_t)t * Dc + d] = (__bf16)((xv[i] - mu) * rstd);
  }
  if (lane == 0) { int pos = atomicAdd(&counts[s0], 1); lists[s0 * Tc + pos] = t; }
  if (lane == 1) { int pos = atomicAdd(&counts[s1], 1); lists[s1 * Tc + pos] = t; }
}

// ---------------------------------------------------------------------------
// Kernel 3: fused expert FFN, async-pipelined.
// ---------------------------------------------------------------------------
__global__ void __launch_bounds__(256)
moe_gemm_kernel(const __bf16* __restrict__ w1bT,   // [E][H][D] bf16 (N-major)
                const __bf16* __restrict__ w2bT,   // [E][D][H] bf16 (N-major)
                const __bf16* __restrict__ xnb,    // [T][D] bf16
                const float* __restrict__ comb,    // [T][E]
                const int* __restrict__ counts,    // [E]
                const int* __restrict__ lists,     // [E][T]
                const float* __restrict__ ln_scale,// [E][D]
                const float* __restrict__ ln_bias, // [E][D]
                const float* __restrict__ gb1,     // [E][H]
                const float* __restrict__ gb2,     // [E][D]
                float* __restrict__ out) {         // [T][D]
  extern __shared__ __align__(16) char smem_raw[];
  __bf16* sh  = (__bf16*)smem_raw;        // [BM][HS]      h tile
  __bf16* sB1 = sh  + BM * HS;            // 2 x [BH][B1S] w1 tiles
  __bf16* sH1 = sB1 + 2 * B1_TILE;        // [BM][H1S]     gelu chunk
  __bf16* sB2 = sH1 + BM * H1S;           // 2 x [Dc][B2S] w2 tiles

  const int tilesPerE = Tc / BM;
  const int e    = blockIdx.x / tilesPerE;
  const int row0 = (blockIdx.x % tilesPerE) * BM;
  const int M    = counts[e];
  if (row0 >= M) return;
  const int rows = (M - row0 < BM) ? (M - row0) : BM;
  const int tid  = threadIdx.x;

  // ---- stage h tile: gather routed tokens, apply per-expert LN scale/bias
  {
    int r  = tid >> 3;
    int c0 = (tid & 7) * (Dc / 8);
    int t = (r < rows) ? lists[e * Tc + row0 + r] : -1;
    const float* sc = ln_scale + (size_t)e * Dc;
    const float* bi = ln_bias  + (size_t)e * Dc;
#pragma unroll 4
    for (int j = 0; j < Dc / 8; ++j) {
      int d = c0 + j;
      float v = 0.f;
      if (t >= 0) v = (float)xnb[(size_t)t * Dc + d] * sc[d] + bi[d];
      sh[r * HS + d] = (__bf16)v;
    }
  }

  const int wv    = tid >> 5;
  const int lane  = tid & 31;
  const int rg    = wv >> 2;      // 0..1 (16-row group)
  const int cg    = wv & 3;       // 0..3 (column group)
  const int lm    = lane & 15;
  const int lhalf = lane >> 4;

  const v8f vzero = {0.f, 0.f, 0.f, 0.f, 0.f, 0.f, 0.f, 0.f};
  v8f acc2[16];
#pragma unroll
  for (int nt = 0; nt < 16; ++nt) acc2[nt] = vzero;

  const unsigned long long g1 =
      (unsigned long long)(uintptr_t)(w1bT + ((size_t)e * Hc) * Dc);
  const unsigned long long g2 =
      (unsigned long long)(uintptr_t)(w2bT + (size_t)e * Dc * Hc);

  for (int hc = 0; hc < Hc; hc += BH) {
    // ============ GEMM1: h1 = h[32x1024] @ w1[:, hc:hc+256] ===============
    v8f acc1[4];
#pragma unroll
    for (int nt = 0; nt < 4; ++nt) acc1[nt] = vzero;

    const unsigned long long g1c = g1 + (size_t)hc * Dc * sizeof(__bf16);
    issue_b1_tile(g1c, sB1, tid, 0);                       // prologue
    for (int kt = 0; kt < Dc / BK1; ++kt) {
      wait_async0();          // own tile-kt copies have landed in LDS
      __syncthreads();        // everyone's have; prior compute done
      if (kt + 1 < Dc / BK1)
        issue_b1_tile(g1c, sB1 + ((kt + 1) & 1) * B1_TILE, tid,
                      (kt + 1) * BK1);
      const __bf16* buf = sB1 + (kt & 1) * B1_TILE;
#pragma unroll
      for (int s = 0; s < 2; ++s) {                        // 2 K-steps of 32
        v16bf A = frag_a(sh + (rg * 16 + lm) * HS + kt * BK1 + s * 32, lhalf);
#pragma unroll
        for (int nt = 0; nt < 4; ++nt) {
          v16bf B = frag_b(buf + (cg * 64 + nt * 16 + lm) * B1S + s * 32,
                           lhalf);
          acc1[nt] = wmma_bf16(A, B, acc1[nt]);
        }
      }
    }
    // ---- epilogue1: + b1, exact GELU, bf16 chunk to LDS (own region)
#pragma unroll
    for (int nt = 0; nt < 4; ++nt) {
      int n = cg * 64 + nt * 16 + lm;
      float bb = gb1[(size_t)e * Hc + hc + n];
#pragma unroll
      for (int r = 0; r < 8; ++r) {
        int m = r + 8 * lhalf;
        float v = acc1[nt][r] + bb;
        float g = 0.5f * v * (1.f + erff(v * 0.70710678118654752f));
        sH1[(rg * 16 + m) * H1S + n] = (__bf16)g;
      }
    }
    // ============ GEMM2: y += h1[32x256] @ w2[hc:hc+256, :] ===============
    issue_b2_tile(g2, sB2, tid, hc);                       // prologue
    for (int kt = 0; kt < BH / BK2; ++kt) {
      wait_async0();
      __syncthreads();        // also orders epilogue1 stores vs A-frag loads
      if (kt + 1 < BH / BK2)
        issue_b2_tile(g2, sB2 + ((kt + 1) & 1) * B2_TILE, tid,
                      hc + (kt + 1) * BK2);
      const __bf16* buf = sB2 + (kt & 1) * B2_TILE;
      v16bf A = frag_a(sH1 + (rg * 16 + lm) * H1S + kt * BK2, lhalf);
#pragma unroll
      for (int nt = 0; nt < 16; ++nt) {
        v16bf B = frag_b(buf + (cg * 256 + nt * 16 + lm) * B2S, lhalf);
        acc2[nt] = wmma_bf16(A, B, acc2[nt]);
      }
    }
  }
  __syncthreads();

  // ---- epilogue2: out[t] += comb[t][e] * (y + b2)
  int   trow[8];
  float cw[8];
#pragma unroll
  for (int r = 0; r < 8; ++r) {
    int m = rg * 16 + r + 8 * lhalf;
    if (m < rows) {
      int t = lists[e * Tc + row0 + m];
      trow[r] = t;
      cw[r]   = comb[(size_t)t * Ec + e];
    } else {
      trow[r] = -1;
      cw[r]   = 0.f;
    }
  }
#pragma unroll
  for (int nt = 0; nt < 16; ++nt) {
    int n = cg * 256 + nt * 16 + lm;
    float bb = gb2[(size_t)e * Dc + n];
#pragma unroll
    for (int r = 0; r < 8; ++r) {
      if (trow[r] >= 0)
        atomicAdd(&out[(size_t)trow[r] * Dc + n], cw[r] * (acc2[nt][r] + bb));
    }
  }
}

// ---------------------------------------------------------------------------
// Launcher.  Workspace: w1bT 64MB | w2bT 64MB | xnb 16MB | comb 256KB |
// counts 256B | lists 256KB  (~151.5 MB total).
// ---------------------------------------------------------------------------
extern "C" void kernel_launch(void* const* d_in, const int* /*in_sizes*/, int /*n_in*/,
                              void* d_out, int /*out_size*/, void* d_ws, size_t /*ws_size*/,
                              hipStream_t stream) {
  const float* x   = (const float*)d_in[0];
  const float* gw  = (const float*)d_in[1];
  const float* ls  = (const float*)d_in[2];
  const float* lb  = (const float*)d_in[3];
  const float* w1  = (const float*)d_in[4];
  const float* b1  = (const float*)d_in[5];
  const float* w2  = (const float*)d_in[6];
  const float* b2  = (const float*)d_in[7];
  float* out = (float*)d_out;

  const size_t NW = (size_t)Ec * Dc * Hc;
  __bf16* w1bT = (__bf16*)d_ws;
  __bf16* w2bT = w1bT + NW;
  __bf16* xnb  = w2bT + NW;
  float*  comb = (float*)(xnb + (size_t)Tc * Dc);
  int* counts  = (int*)(comb + (size_t)Tc * Ec);
  int* lists   = counts + 64;

  init_kernel<<<(Tc * Dc) / 1024, 256, 0, stream>>>(out, counts);
  transpose_conv_kernel<<<Ec * (Dc / 32) * (Hc / 32), 256, 0, stream>>>(w1, w1bT, Dc, Hc);
  transpose_conv_kernel<<<Ec * (Hc / 32) * (Dc / 32), 256, 0, stream>>>(w2, w2bT, Hc, Dc);
  gate_ln_kernel<<<Tc / 8, 256, 0, stream>>>(x, gw, xnb, comb, counts, lists);
  moe_gemm_kernel<<<Ec * (Tc / BM), 256, SMEM_BYTES, stream>>>(
      w1bT, w2bT, xnb, comb, counts, lists, ls, lb, b1, b2, out);
}
